// GCN1_46024869544124
// MI455X (gfx1250) — matrix operand order repaired and checked by reading, not compiled, not run on hardware
//
#include <hip/hip_runtime.h>
#include <hip/hip_bf16.h>
#include <math.h>

typedef __bf16 bf16_t;
typedef __attribute__((ext_vector_type(16))) __bf16 v16bf;
typedef __attribute__((ext_vector_type(8)))  __bf16 v8bf;
typedef __attribute__((ext_vector_type(8)))  float  v8f;

#define GSTRIDE (gridDim.x * blockDim.x)
#define GTID    (blockIdx.x * blockDim.x + threadIdx.x)

// ---------------- elementwise / graph-norm kernels ----------------

__global__ void k_zero_f32(float* __restrict__ p, int n) {
  for (int i = GTID; i < n; i += GSTRIDE) p[i] = 0.0f;
}

__global__ void k_deg(const int* __restrict__ dst, const float* __restrict__ w,
                      float* __restrict__ deg, int E) {
  for (int e = GTID; e < E; e += GSTRIDE)
    atomicAdd(&deg[dst[e]], fabsf(w[e]));
}

__global__ void k_dis(float* __restrict__ d, int n) {
  for (int i = GTID; i < n; i += GSTRIDE) d[i] = rsqrtf(d[i] + 1.0f);
}

__global__ void k_norme(const int* __restrict__ src, const int* __restrict__ dst,
                        const float* __restrict__ w, const float* __restrict__ dis,
                        float* __restrict__ ne, int E) {
  for (int e = GTID; e < E; e += GSTRIDE)
    ne[e] = dis[src[e]] * fabsf(w[e]) * dis[dst[e]];
}

__global__ void k_f32_to_bf16(const float* __restrict__ x, bf16_t* __restrict__ y, int n) {
  for (int i = GTID; i < n; i += GSTRIDE) y[i] = (bf16_t)x[i];
}

// ---------------- weight packing into WMMA B-matrix lane layout ----------------
// Layout: Wp[((nt*KS + ks)*32 + lane)*16 + h], where for 16-bit B (32x16 per k-step):
//   lanes 0-15 hold K=ks*32+0..7 (h<8) and 16..23 (h>=8),
//   lanes 16-31 hold K=ks*32+8..15 and 24..31, column N = nt*16 + (lane&15).
__global__ void k_pack_w(const float* __restrict__ W, bf16_t* __restrict__ Wp,
                         int Fin, int FoutLog, int Fp) {
  int KS = Fin >> 5;
  int total = (Fp >> 4) * KS * 32 * 16;
  for (int t = GTID; t < total; t += GSTRIDE) {
    int h    = t & 15;
    int lane = (t >> 4) & 31;
    int blk  = t >> 9;           // nt*KS + ks
    int ks   = blk % KS;
    int nt   = blk / KS;
    int k = ks * 32 + ((lane >> 4) << 3) + (h & 7) + ((h >> 3) << 4);
    int n = nt * 16 + (lane & 15);
    float v = (n < FoutLog) ? W[k * FoutLog + n] : 0.0f;
    Wp[t] = (bf16_t)v;
  }
}

// ---------------- bf16 WMMA GEMM: xw[M,Fp] = A[M,Fin] @ W ----------------
// One wave computes one 16x16 output tile; K-loop of v_wmma_f32_16x16x32_bf16.
__global__ void k_gemm_wmma(const bf16_t* __restrict__ A, const bf16_t* __restrict__ Wp,
                            float* __restrict__ C, int M, int Fin, int Fp) {
  int NT = Fp >> 4, KS = Fin >> 5;
  int lane   = threadIdx.x & 31;
  int wave   = (blockIdx.x * blockDim.x + threadIdx.x) >> 5;
  int nWaves = (gridDim.x * blockDim.x) >> 5;
  int tiles  = (M >> 4) * NT;
  int mrow  = lane & 15;
  int koffA = (lane >> 4) * 8;   // 16-bit A layout: half-wave K split 0..7 / 8..15 (+16)

  for (int t = wave; t < tiles; t += nWaves) {
    int mt = t / NT, nt = t % NT;
    const bf16_t* pa = A + (size_t)(mt * 16 + mrow) * Fin + koffA;
    const bf16_t* pb = Wp + ((size_t)nt * KS * 32 + lane) * 16;
    v8f c = {};
    for (int ks = 0; ks < KS; ++ks) {
      v8bf alo = *(const v8bf*)(pa);        // K = koffA .. koffA+7   (VGPRs 0-3)
      v8bf ahi = *(const v8bf*)(pa + 16);   // K = koffA+16 .. +23    (VGPRs 4-7)
      v16bf a;
#pragma unroll
      for (int i = 0; i < 8; ++i) { a[i] = alo[i]; a[8 + i] = ahi[i]; }
      v16bf b = *(const v16bf*)(pb);
      c = __builtin_amdgcn_wmma_f32_16x16x32_bf16(false, a, false, b,
                                                  (short)0, c, false, false);
      pa += 32;
      pb += 32 * 16;
    }
    // C/D layout: VGPR v, lanes 0-15 -> M=v, lanes 16-31 -> M=v+8, N=lane&15
    int ncol  = nt * 16 + (lane & 15);
    int rbase = mt * 16 + 8 * (lane >> 4);
#pragma unroll
    for (int v = 0; v < 8; ++v)
      C[(size_t)(rbase + v) * Fp + ncol] = c[v];
  }
}

// ---------------- aggregation ----------------

__global__ void k_init_acc(const float* __restrict__ xw, const float* __restrict__ dis,
                           const float* __restrict__ bias, float* __restrict__ acc,
                           int N, int Fp, int Flog) {
  int total = N * Fp;
  for (int i = GTID; i < total; i += GSTRIDE) {
    int n = i / Fp, f = i - n * Fp;
    float sn = dis[n]; sn *= sn;                     // self-loop norm D^-1
    float bv = (f < Flog) ? bias[f] : 0.0f;
    acc[i] = sn * xw[i] + bv;
  }
}

__global__ void k_scatter(const int* __restrict__ src, const int* __restrict__ dst,
                          const float* __restrict__ ne, const float* __restrict__ xw,
                          float* __restrict__ acc, int E, int Fp) {
  int lane = threadIdx.x & 31;
  int wave = (blockIdx.x * blockDim.x + threadIdx.x) >> 5;
  int nW   = (gridDim.x * blockDim.x) >> 5;
  for (int e = wave; e < E; e += nW) {
    int e2 = e + nW;
    if (e2 < E)
      __builtin_prefetch(xw + (size_t)src[e2] * Fp + lane, 0, 1);
    int s = src[e], d = dst[e];
    float w = ne[e];
    const float* ps = xw + (size_t)s * Fp;
    float* pd = acc + (size_t)d * Fp;
    for (int f = lane; f < Fp; f += 32)
      atomicAdd(&pd[f], w * ps[f]);
  }
}

__global__ void k_relu_bf16(const float* __restrict__ acc, bf16_t* __restrict__ hb, int n) {
  for (int i = GTID; i < n; i += GSTRIDE) hb[i] = (bf16_t)fmaxf(acc[i], 0.0f);
}

// ---------------- pooling + head ----------------

__global__ void k_pool(const float* __restrict__ acc, const int* __restrict__ batch,
                       float* __restrict__ sums, float* __restrict__ cnt,
                       int N, int Fp, int Flog) {
  int total = N * 32;
  for (int t = GTID; t < total; t += GSTRIDE) {
    int n = t >> 5, f = t & 31;
    int g = batch[n];
    if (f < Flog)
      atomicAdd(&sums[g * Flog + f], fmaxf(acc[(size_t)n * Fp + f], 0.0f));
    else if (f == 31)
      atomicAdd(&cnt[g], 1.0f);
  }
}

__global__ void k_head(const float* __restrict__ sums, const float* __restrict__ cnt,
                       const float* __restrict__ Wf, const float* __restrict__ bfv,
                       float* __restrict__ out, int G) {
  int g = GTID;
  if (g >= G) return;
  float inv = 1.0f / fmaxf(cnt[g], 1.0f);
  float logit[10];
#pragma unroll
  for (int c = 0; c < 10; ++c) logit[c] = bfv[c];
  for (int f = 0; f < 30; ++f) {
    float p = sums[g * 30 + f] * inv;
#pragma unroll
    for (int c = 0; c < 10; ++c) logit[c] += p * Wf[f * 10 + c];
  }
  float mx = logit[0];
#pragma unroll
  for (int c = 1; c < 10; ++c) mx = fmaxf(mx, logit[c]);
  float s = 0.0f;
#pragma unroll
  for (int c = 0; c < 10; ++c) { logit[c] = __expf(logit[c] - mx); s += logit[c]; }
  float r = 1.0f / s;
#pragma unroll
  for (int c = 0; c < 10; ++c) out[g * 10 + c] = logit[c] * r;
}

// ---------------- host launcher ----------------

extern "C" void kernel_launch(void* const* d_in, const int* in_sizes, int n_in,
                              void* d_out, int out_size, void* d_ws, size_t ws_size,
                              hipStream_t stream) {
  const float* x     = (const float*)d_in[0];
  const int*   ei    = (const int*)d_in[1];
  const float* ew    = (const float*)d_in[2];
  const int*   batch = (const int*)d_in[3];
  const float* W1 = (const float*)d_in[4];  const float* b1 = (const float*)d_in[5];
  const float* W2 = (const float*)d_in[6];  const float* b2 = (const float*)d_in[7];
  const float* W3 = (const float*)d_in[8];  const float* b3 = (const float*)d_in[9];
  const float* Wf = (const float*)d_in[10]; const float* bf_ = (const float*)d_in[11];
  float* out = (float*)d_out;

  const int N = in_sizes[0] / 128;   // 100000
  const int E = in_sizes[1] / 2;     // 3200000
  const int G = 256;
  const int* src = ei;
  const int* dst = ei + E;

  // carve workspace (256B-aligned regions)
  char* p = (char*)d_ws;
  auto carve = [&](size_t bytes) -> char* {
    char* r = p;
    p += (bytes + 255) & ~(size_t)255;
    return r;
  };
  float*  dis  = (float*) carve((size_t)N * 4);
  float*  ne   = (float*) carve((size_t)E * 4);
  bf16_t* hb   = (bf16_t*)carve((size_t)N * 128 * 2);
  float*  xw   = (float*) carve((size_t)N * 96 * 4);
  float*  acc  = (float*) carve((size_t)N * 96 * 4);
  bf16_t* Wp   = (bf16_t*)carve((size_t)128 * 96 * 2);
  float*  sums = (float*) carve((size_t)G * 30 * 4);
  float*  cnt  = (float*) carve((size_t)G * 4);

  const int TB = 256;
  auto blocks = [&](long long work) -> int {
    long long b = (work + TB - 1) / TB;
    return (int)(b > 65535 ? 65535 : (b < 1 ? 1 : b));
  };

  // normalization
  k_zero_f32<<<blocks(N), TB, 0, stream>>>(dis, N);
  k_deg     <<<blocks(E), TB, 0, stream>>>(dst, ew, dis, E);
  k_dis     <<<blocks(N), TB, 0, stream>>>(dis, N);
  k_norme   <<<blocks(E), TB, 0, stream>>>(src, dst, ew, dis, ne, E);
  k_f32_to_bf16<<<blocks((long long)N * 128), TB, 0, stream>>>(x, hb, N * 128);

  struct Layer { const float* W; const float* b; int Fin, Fp, Flog; };
  Layer L[3] = { {W1, b1, 128, 96, 96},
                 {W2, b2,  96, 96, 96},
                 {W3, b3,  96, 32, 30} };

  for (int l = 0; l < 3; ++l) {
    int Fin = L[l].Fin, Fp = L[l].Fp, Flog = L[l].Flog;
    int NT = Fp / 16, KS = Fin / 32;
    k_pack_w<<<blocks((long long)NT * KS * 512), TB, 0, stream>>>(L[l].W, Wp, Fin, Flog, Fp);
    int tiles = (N / 16) * NT;
    int gblk = (tiles + 7) / 8;           // 8 waves per 256-thread block, 1 tile/wave
    if (gblk > 65535) gblk = 65535;
    k_gemm_wmma<<<gblk, TB, 0, stream>>>(hb, Wp, xw, N, Fin, Fp);
    k_init_acc<<<blocks((long long)N * Fp), TB, 0, stream>>>(xw, dis, L[l].b, acc, N, Fp, Flog);
    k_scatter<<<4096, TB, 0, stream>>>(src, dst, ne, xw, acc, E, Fp);
    if (l < 2)
      k_relu_bf16<<<blocks((long long)N * Fp), TB, 0, stream>>>(acc, hb, N * Fp);
  }

  // global mean pool + classifier head
  k_zero_f32<<<blocks(G * 30), TB, 0, stream>>>(sums, G * 30);
  k_zero_f32<<<1, TB, 0, stream>>>(cnt, G);
  k_pool<<<blocks((long long)N * 32), TB, 0, stream>>>(acc, batch, sums, cnt, N, 32, 30);
  k_head<<<1, TB, 0, stream>>>(sums, cnt, Wf, bf_, out, G);
}
